// GNN_1803886264420
// MI455X (gfx1250) — compile-verified, matrix operand order checked by reading, not run on hardware
//
#include <hip/hip_runtime.h>
#include <hip/hip_bf16.h>
#include <math.h>

// ---------------------------------------------------------------------------
// GATv2 (2 layers) for MI455X / gfx1250 (wave32, WMMA bf16 16x16x32).
//
// - All GEMM work via v_wmma_f32_16x16x32_bf16.
// - The big edge GEMM (edge_attr @ We, 26 GFLOP/layer) is FUSED into the
//   logits pass so the [E,256] intermediate (819 MB) never touches HBM.
// - B fragments are staged in LDS once per 4-wave workgroup via
//   global_load_async_to_lds_b128 + s_wait_asynccnt and re-used by all
//   waves, feeding WMMA from ds_load_b128 instead of per-wave VMEM.
// - xl/xr/out (51 MB each) stay L2-resident (192 MB L2): per-edge gathers
//   and scatter atomics are L2 traffic. HBM ~250 MB/layer -> ~11 us floor.
// - Store epilogues branch once on a uniform "tile fully in range" test so
//   the common path is straight-line stores (no per-element exec masking).
// ---------------------------------------------------------------------------

typedef __attribute__((ext_vector_type(16))) __bf16 v16bf;
typedef __attribute__((ext_vector_type(8)))  float  v8f;

#define HC    256   // heads * hidden
#define HEADS 4

#if defined(__AMDGCN__) && __has_builtin(__builtin_amdgcn_global_load_async_to_lds_b128) && __has_builtin(__builtin_amdgcn_s_wait_asynccnt)
#define HAVE_ASYNC_LDS 1
typedef int v4i __attribute__((ext_vector_type(4)));
typedef __attribute__((address_space(1))) v4i* gv4i_p;   // global int4*
typedef __attribute__((address_space(3))) v4i* lv4i_p;   // LDS int4*
#else
#define HAVE_ASYNC_LDS 0
#endif

// 16-bit A-matrix 16x32 layout (ISA 7.12.2): lane<16 holds K {0..7,16..23},
// lane>=16 holds K {8..15,24..31}; VGPR p holds pair (2p,2p+1) within group.
__device__ __forceinline__ int kpair(int p, int lane) {
  return ((p < 4) ? 2 * p : 16 + 2 * (p - 4)) + ((lane & 16) ? 8 : 0);
}

// Sign-aware float atomic max via integer atomics (works with -INF init).
__device__ __forceinline__ void atomicMaxF(float* addr, float value) {
  if (!(value == value)) return;
  if (value >= 0.0f)
    atomicMax((int*)addr, __float_as_int(value));
  else
    atomicMin((unsigned int*)addr, __float_as_uint(value));
}

// Cooperative global->LDS stage of nvec 16-byte vectors (whole workgroup).
__device__ __forceinline__ void stage_to_lds(const uint4* __restrict__ g,
                                             uint4* l, int nvec) {
#if HAVE_ASYNC_LDS
  for (int i = threadIdx.x; i < nvec; i += blockDim.x)
    __builtin_amdgcn_global_load_async_to_lds_b128((gv4i_p)(g + i),
                                                   (lv4i_p)(l + i), 0, 0);
  __builtin_amdgcn_s_wait_asynccnt(0);
#else
  for (int i = threadIdx.x; i < nvec; i += blockDim.x) l[i] = g[i];
#endif
  __syncthreads();
}

// ---------------------------------------------------------------------------
// Repack W[K x 256] (f32, row-major) into bf16 WMMA B-fragments.
// Fragment (ntile,ktile) stored at ((ntile*nkt + ktile)*32 + lane)*16 bf16,
// lane<16 -> column ntile*16+lane, K pairs mirroring the A layout.
// ---------------------------------------------------------------------------
__global__ __launch_bounds__(32) void pack_b_kernel(const float* __restrict__ W,
                                                    __bf16* __restrict__ out) {
  const int lane = threadIdx.x;
  const int nt   = blockIdx.x;          // 0..15 column tile
  const int kt   = blockIdx.y;          // 0..K/32-1
  const int nkt  = gridDim.y;
  const int n    = nt * 16 + (lane & 15);
  __bf16* o = out + ((size_t)(nt * nkt + kt) * 32 + lane) * 16;
#pragma unroll
  for (int p = 0; p < 8; ++p) {
    int kk = kt * 32 + kpair(p, lane);
    o[2 * p]     = (__bf16)W[(size_t)kk * HC + n];
    o[2 * p + 1] = (__bf16)W[(size_t)(kk + 1) * HC + n];
  }
}

// ---------------------------------------------------------------------------
// C[M x 256] = bf16(X[M x K]) @ Wp + bias.
// 4 waves/WG; each wave owns a 16-row tile; the 64-column B panel
// (4 col-tiles x nkt k-tiles, contiguous in Wp) is staged in LDS once.
// ---------------------------------------------------------------------------
__global__ __launch_bounds__(128) void node_gemm_kernel(
    const float* __restrict__ X, const __bf16* __restrict__ Wp,
    const float* __restrict__ bias, float* __restrict__ C, int M, int K) {
  __shared__ __align__(64) __bf16 ldsB[4 * 8 * 512];   // up to K=256: 32 KB
  const int nkt = K >> 5;
  const int ng  = blockIdx.y * 4;                      // 4 col-tiles = 64 cols

  stage_to_lds((const uint4*)(Wp + (size_t)ng * nkt * 512),
               (uint4*)ldsB, nkt * 256 /* = 4*nkt*1024B / 16 */);

  const int lane = threadIdx.x & 31;
  const int wv   = threadIdx.x >> 5;
  const int m0   = (blockIdx.x * 4 + wv) * 16;
  const int half = (lane & 16) ? 8 : 0;
  v8f acc[4] = {};

  int rowA = m0 + (lane & 15);
  if (rowA > M - 1) rowA = M - 1;
  const float* arow = X + (size_t)rowA * K;

  for (int kt = 0; kt < nkt; ++kt) {
    v16bf a;
#pragma unroll
    for (int p = 0; p < 8; ++p) {
      int kk = kt * 32 + kpair(p, lane);
      a[2 * p]     = (__bf16)arow[kk];
      a[2 * p + 1] = (__bf16)arow[kk + 1];
    }
#pragma unroll
    for (int ct = 0; ct < 4; ++ct) {
      v16bf b = *(const v16bf*)(ldsB + ((size_t)(ct * nkt + kt) * 32 + lane) * 16);
      acc[ct] = __builtin_amdgcn_wmma_f32_16x16x32_bf16(
          false, a, false, b, (short)0, acc[ct], false, false);
    }
  }

  const int colb = lane & 15;
  if (m0 + 16 <= M) {                 // uniform fast path: straight-line stores
#pragma unroll
    for (int ct = 0; ct < 4; ++ct) {
      const int col = (ng + ct) * 16 + colb;
      const float bv = bias[col];
      float* crow = C + (size_t)(m0 + half) * HC + col;
#pragma unroll
      for (int v = 0; v < 8; ++v) crow[(size_t)v * HC] = acc[ct][v] + bv;
    }
  } else {                            // ragged tail (last workgroup only)
#pragma unroll
    for (int ct = 0; ct < 4; ++ct) {
      const int col = (ng + ct) * 16 + colb;
      const float bv = bias[col];
#pragma unroll
      for (int v = 0; v < 8; ++v) {
        const int row = m0 + half + v;
        if (row < M) C[(size_t)row * HC + col] = acc[ct][v] + bv;
      }
    }
  }
}

// ---------------------------------------------------------------------------
// Fused edge pass: per 16-edge tile compute ee = edge_attr@We via WMMA
// (K=64 -> two k-steps), add gathered xl[src]+xr[dst] on the C-fragment
// layout, LeakyReLU(0.2), scale by att, reduce per head with shfl_xor over
// 16-lane halves -> logits[E,4]; atomic max into nodeMax[N,4].
// 4 waves/WG (4 edge tiles); the whole packed We (32 KB) is staged in LDS.
// ---------------------------------------------------------------------------
__global__ __launch_bounds__(128) void edge_logits_kernel(
    const float* __restrict__ ea, const int* __restrict__ src,
    const int* __restrict__ dst, const __bf16* __restrict__ Wep,
    const float* __restrict__ xl, const float* __restrict__ xr,
    const float* __restrict__ att, float* __restrict__ logits,
    float* __restrict__ nodeMax, int E) {
  __shared__ __align__(64) __bf16 ldsWe[16 * 2 * 512];   // 32 KB
  stage_to_lds((const uint4*)Wep, (uint4*)ldsWe, 2048);

  const int  lane = threadIdx.x & 31;
  const int  wv   = threadIdx.x >> 5;
  const long e0   = ((long)blockIdx.x * 4 + wv) * 16;
  const int  half = (lane & 16) ? 8 : 0;
  const bool full = (e0 + 16 <= E);   // uniform per wave

  // A fragments (16 edges x 64), f32 -> bf16 on the fly.
  long eA = e0 + (lane & 15);
  if (eA > E - 1) eA = E - 1;
  const float* arow = ea + eA * 64;
  v16bf a0, a1;
#pragma unroll
  for (int p = 0; p < 8; ++p) {
    const int kb = kpair(p, lane);
    a0[2 * p]     = (__bf16)arow[kb];
    a0[2 * p + 1] = (__bf16)arow[kb + 1];
    a1[2 * p]     = (__bf16)arow[32 + kb];
    a1[2 * p + 1] = (__bf16)arow[32 + kb + 1];
  }

  int sV[8], dV[8];
#pragma unroll
  for (int v = 0; v < 8; ++v) {
    long e = e0 + half + v;
    if (e > E - 1) e = E - 1;
    sV[v] = src[e];
    dV[v] = dst[e];
  }
  // Warm the gather rows (global_prefetch_b8); one lane per half suffices.
  if ((lane & 15) == 0) {
#pragma unroll
    for (int v = 0; v < 8; ++v) {
      __builtin_prefetch(xl + (size_t)sV[v] * HC, 0, 0);
      __builtin_prefetch(xr + (size_t)dV[v] * HC, 0, 0);
    }
  }

  float accl[8];
#pragma unroll
  for (int v = 0; v < 8; ++v) accl[v] = 0.0f;
  const int colb = lane & 15;

  for (int n = 0; n < 16; ++n) {                 // 16 column tiles of 16
    const __bf16* bp = ldsWe + ((size_t)(n * 2) * 32 + lane) * 16;
    v16bf b0 = *(const v16bf*)bp;
    v16bf b1 = *(const v16bf*)(bp + 512);
    v8f c = {};
    c = __builtin_amdgcn_wmma_f32_16x16x32_bf16(false, a0, false, b0, (short)0, c, false, false);
    c = __builtin_amdgcn_wmma_f32_16x16x32_bf16(false, a1, false, b1, (short)0, c, false, false);

    const int col = n * 16 + colb;
    const float aw = att[col];                   // att[h*64+c] == att[col]
#pragma unroll
    for (int v = 0; v < 8; ++v) {
      float val = c[v] + xl[(size_t)sV[v] * HC + col] + xr[(size_t)dV[v] * HC + col];
      val = (val > 0.0f) ? val : 0.2f * val;     // leaky_relu(0.2)
      accl[v] += val * aw;
    }

    if ((n & 3) == 3) {                          // head h = n>>2 finished
      const int h = n >> 2;
#pragma unroll
      for (int v = 0; v < 8; ++v) {
        float r = accl[v];
        r += __shfl_xor(r, 1, 32);
        r += __shfl_xor(r, 2, 32);
        r += __shfl_xor(r, 4, 32);
        r += __shfl_xor(r, 8, 32);               // reduce within 16-lane half
        accl[v] = r;
      }
      if ((lane & 15) == 0) {                    // lane 0 -> edges 0..7, lane 16 -> 8..15
        float* lp = logits + (e0 + half) * HEADS + h;
        if (full) {
#pragma unroll
          for (int v = 0; v < 8; ++v) {
            lp[v * HEADS] = accl[v];
            atomicMaxF(&nodeMax[(size_t)dV[v] * HEADS + h], accl[v]);
          }
        } else {
#pragma unroll
          for (int v = 0; v < 8; ++v) {
            if (e0 + half + v < E) {
              lp[v * HEADS] = accl[v];
              atomicMaxF(&nodeMax[(size_t)dV[v] * HEADS + h], accl[v]);
            }
          }
        }
      }
#pragma unroll
      for (int v = 0; v < 8; ++v) accl[v] = 0.0f;
    }
  }
}

// ---------------------------------------------------------------------------
__global__ void init_node_stats_kernel(float* __restrict__ nmax,
                                       float* __restrict__ nsum, int n) {
  int i = blockIdx.x * blockDim.x + threadIdx.x;
  if (i < n) { nmax[i] = -INFINITY; nsum[i] = 0.0f; }
}

__global__ void init_out_bias_kernel(float* __restrict__ out,
                                     const float* __restrict__ bias, int total) {
  int i = blockIdx.x * blockDim.x + threadIdx.x;
  if (i < total) out[i] = bias[i & (HC - 1)];
}

// w = exp(logit - max[dst]); accumulate per-node sums.
__global__ void edge_exp_kernel(const int* __restrict__ dst,
                                float* __restrict__ logw,
                                const float* __restrict__ nmax,
                                float* __restrict__ nsum, int total) {
  int i = blockIdx.x * blockDim.x + threadIdx.x;
  if (i >= total) return;
  const int e = i >> 2, h = i & 3;
  const int d = dst[e];
  float m = nmax[(size_t)d * HEADS + h];
  if (!(m > -1e30f)) m = 0.0f;                   // reference: non-finite -> 0
  const float w = __expf(logw[i] - m);
  logw[i] = w;                                   // in-place: logits -> exp weights
  atomicAdd(&nsum[(size_t)d * HEADS + h], w);
}

// nsum -> 1/(nsum+eps) so the hot aggregation pass multiplies instead of divides.
__global__ void node_recip_kernel(float* __restrict__ nsum, int n) {
  int i = blockIdx.x * blockDim.x + threadIdx.x;
  if (i < n) nsum[i] = 1.0f / (nsum[i] + 1e-16f);
}

// out[dst] += alpha * xl[src]; 64 threads/edge, float4 each, f32 atomics (L2).
__global__ void edge_aggr_kernel(const int* __restrict__ src,
                                 const int* __restrict__ dst,
                                 const float* __restrict__ expw,
                                 const float* __restrict__ nsumR,
                                 const float* __restrict__ xl,
                                 float* __restrict__ out, long total) {
  long i = (long)blockIdx.x * blockDim.x + threadIdx.x;
  if (i >= total) return;
  const long e = i >> 6;
  const int  q = (int)(i & 63);
  const int  h = q >> 4;                          // 16 float4's per head
  const int  s = src[e], d = dst[e];
  const float alpha = expw[e * HEADS + h] * nsumR[(size_t)d * HEADS + h];
  const float4 v = *(const float4*)(xl + (size_t)s * HC + q * 4);
  float* o = out + (size_t)d * HC + q * 4;
  atomicAdd(o + 0, alpha * v.x);
  atomicAdd(o + 1, alpha * v.y);
  atomicAdd(o + 2, alpha * v.z);
  atomicAdd(o + 3, alpha * v.w);
}

// ---------------------------------------------------------------------------
extern "C" void kernel_launch(void* const* d_in, const int* in_sizes, int n_in,
                              void* d_out, int out_size, void* d_ws, size_t ws_size,
                              hipStream_t stream) {
  const float* x    = (const float*)d_in[0];
  const int*   ei   = (const int*)  d_in[1];
  const float* ea   = (const float*)d_in[2];
  const float* Wl1  = (const float*)d_in[3];
  const float* bl1  = (const float*)d_in[4];
  const float* Wr1  = (const float*)d_in[5];
  const float* br1  = (const float*)d_in[6];
  const float* We1  = (const float*)d_in[7];
  const float* att1 = (const float*)d_in[8];
  const float* bias1= (const float*)d_in[9];
  const float* Wl2  = (const float*)d_in[10];
  const float* bl2  = (const float*)d_in[11];
  const float* Wr2  = (const float*)d_in[12];
  const float* br2  = (const float*)d_in[13];
  const float* We2  = (const float*)d_in[14];
  const float* att2 = (const float*)d_in[15];
  const float* bias2= (const float*)d_in[16];

  const int IN = in_sizes[3] / HC;      // 128
  const int ED = in_sizes[7] / HC;      // 64
  const int Nn = in_sizes[0] / IN;      // 50000
  const int E  = in_sizes[1] / 2;       // 800000
  const int* srcIdx = ei;
  const int* dstIdx = ei + E;

  // Workspace carve-out (256B aligned).
  char* ws = (char*)d_ws;
  size_t cur = 0;
  auto carve = [&](size_t bytes) -> char* {
    char* p = ws + cur;
    cur = (cur + bytes + 255) & ~(size_t)255;
    return p;
  };
  float*  xl   = (float*)carve((size_t)Nn * HC * 4);
  float*  xr   = (float*)carve((size_t)Nn * HC * 4);
  float*  h1   = (float*)carve((size_t)Nn * HC * 4);
  float*  logw = (float*)carve((size_t)E * HEADS * 4);
  float*  nmax = (float*)carve((size_t)Nn * HEADS * 4);
  float*  nsum = (float*)carve((size_t)Nn * HEADS * 4);
  __bf16* Wl1p = (__bf16*)carve((size_t)16 * (IN / 32) * 512 * 2);
  __bf16* Wr1p = (__bf16*)carve((size_t)16 * (IN / 32) * 512 * 2);
  __bf16* We1p = (__bf16*)carve((size_t)16 * (ED / 32) * 512 * 2);
  __bf16* Wl2p = (__bf16*)carve((size_t)16 * (HC / 32) * 512 * 2);
  __bf16* Wr2p = (__bf16*)carve((size_t)16 * (HC / 32) * 512 * 2);
  __bf16* We2p = (__bf16*)carve((size_t)16 * (ED / 32) * 512 * 2);

  // Weight repack (tiny, once per call).
  pack_b_kernel<<<dim3(16, IN / 32), 32, 0, stream>>>(Wl1, Wl1p);
  pack_b_kernel<<<dim3(16, IN / 32), 32, 0, stream>>>(Wr1, Wr1p);
  pack_b_kernel<<<dim3(16, ED / 32), 32, 0, stream>>>(We1, We1p);
  pack_b_kernel<<<dim3(16, HC / 32), 32, 0, stream>>>(Wl2, Wl2p);
  pack_b_kernel<<<dim3(16, HC / 32), 32, 0, stream>>>(Wr2, Wr2p);
  pack_b_kernel<<<dim3(16, ED / 32), 32, 0, stream>>>(We2, We2p);

  const int  mtiles = (Nn + 15) / 16;
  const dim3 ggrid((mtiles + 3) / 4, 4);          // 4 waves/WG, 4 col groups
  const int  eblocks = (E + 63) / 64;             // 4 edge tiles per WG
  const int  nstat  = Nn * HEADS;
  const int  ntot   = Nn * HC;
  const int  etot   = E * HEADS;
  const long atot   = (long)E * 64;

  // ---------------- Layer 1 ----------------
  node_gemm_kernel<<<ggrid, 128, 0, stream>>>(x, Wl1p, bl1, xl, Nn, IN);
  node_gemm_kernel<<<ggrid, 128, 0, stream>>>(x, Wr1p, br1, xr, Nn, IN);
  init_node_stats_kernel<<<(nstat + 255) / 256, 256, 0, stream>>>(nmax, nsum, nstat);
  init_out_bias_kernel<<<(ntot + 255) / 256, 256, 0, stream>>>(h1, bias1, ntot);
  edge_logits_kernel<<<eblocks, 128, 0, stream>>>(ea, srcIdx, dstIdx, We1p, xl, xr,
                                                  att1, logw, nmax, E);
  edge_exp_kernel<<<(etot + 255) / 256, 256, 0, stream>>>(dstIdx, logw, nmax, nsum, etot);
  node_recip_kernel<<<(nstat + 255) / 256, 256, 0, stream>>>(nsum, nstat);
  edge_aggr_kernel<<<(int)((atot + 255) / 256), 256, 0, stream>>>(srcIdx, dstIdx, logw,
                                                                  nsum, xl, h1, atot);

  // ---------------- Layer 2 ----------------
  node_gemm_kernel<<<ggrid, 128, 0, stream>>>(h1, Wl2p, bl2, xl, Nn, HC);
  node_gemm_kernel<<<ggrid, 128, 0, stream>>>(h1, Wr2p, br2, xr, Nn, HC);
  init_node_stats_kernel<<<(nstat + 255) / 256, 256, 0, stream>>>(nmax, nsum, nstat);
  init_out_bias_kernel<<<(ntot + 255) / 256, 256, 0, stream>>>((float*)d_out, bias2, ntot);
  edge_logits_kernel<<<eblocks, 128, 0, stream>>>(ea, srcIdx, dstIdx, We2p, xl, xr,
                                                  att2, logw, nmax, E);
  edge_exp_kernel<<<(etot + 255) / 256, 256, 0, stream>>>(dstIdx, logw, nmax, nsum, etot);
  node_recip_kernel<<<(nstat + 255) / 256, 256, 0, stream>>>(nsum, nstat);
  edge_aggr_kernel<<<(int)((atot + 255) / 256), 256, 0, stream>>>(srcIdx, dstIdx, logw,
                                                                  nsum, xl, (float*)d_out, atot);
}
